// CausalSelfAttention_11029476016603
// MI455X (gfx1250) — matrix operand order repaired and checked by reading, compiled, not run on hardware
//
#include <hip/hip_runtime.h>

typedef __attribute__((ext_vector_type(16))) _Float16 v16h;
typedef __attribute__((ext_vector_type(8)))  _Float16 v8h;
typedef __attribute__((ext_vector_type(4)))  _Float16 v4h;
typedef __attribute__((ext_vector_type(8)))  float    v8f;
typedef __attribute__((ext_vector_type(4)))  int      v4i;

union V16 { v16h v; v8h h[2]; };

// Fixed problem dims
constexpr int CB = 2;        // batch
constexpr int CT = 2048;     // seq len
constexpr int CC = 1024;     // embed
constexpr int CH = 16;       // heads
constexpr int CD = 64;       // head dim

#if defined(__has_builtin)
#if __has_builtin(__builtin_amdgcn_global_load_async_to_lds_b128) && \
    __has_builtin(__builtin_amdgcn_s_wait_asynccnt)
#define USE_ASYNC_LDS 1
#endif
#endif

#if defined(USE_ASYNC_LDS)
typedef __attribute__((address_space(1))) v4i* g4_ptr;
typedef __attribute__((address_space(3))) v4i* l4_ptr;
static __device__ __forceinline__ void async_copy16(const void* g, void* l) {
    __builtin_amdgcn_global_load_async_to_lds_b128((g4_ptr)g, (l4_ptr)l, 0, 0);
}
static __device__ __forceinline__ void async_wait0() {
    __builtin_amdgcn_s_wait_asynccnt(0);
}
#else
static __device__ __forceinline__ void async_copy16(const void* g, void* l) {
    *(v8h*)l = *(const v8h*)g;
}
static __device__ __forceinline__ void async_wait0() {}
#endif

// A-matrix 16x32 f16 fragment (row-major source): lane&15 = row,
// K elems 0..15 contiguous at koff = (lane<16)?0:16
static __device__ __forceinline__ v16h load_fragA(const _Float16* base, int stride, int lane) {
    int r = lane & 15;
    int koff = (lane & 16) ? 16 : 0;
    const _Float16* p = base + r * stride + koff;
    V16 u;
    u.h[0] = *(const v8h*)(p);
    u.h[1] = *(const v8h*)(p + 8);
    return u.v;
}

// B-matrix 32x16 f16 fragment (column n stored row-major at base+n*stride):
// lane&15 = column, K elems 0..15 contiguous at koff = (lane<16)?0:16
static __device__ __forceinline__ v16h load_fragB(const _Float16* base, int stride, int lane) {
    int c = lane & 15;
    int koff = (lane & 16) ? 16 : 0;
    const _Float16* p = base + c * stride + koff;
    V16 u;
    u.h[0] = *(const v8h*)(p);
    u.h[1] = *(const v8h*)(p + 8);
    return u.v;
}

static __device__ __forceinline__ v8f wmma16(v16h a, v16h b, v8f c) {
    return __builtin_amdgcn_wmma_f32_16x16x32_f16(false, a, false, b, (short)0, c, false, false);
}

// ---------------------------------------------------------------------------
// f32 -> f16 elementwise convert (n4 = n/4 float4 units)
// ---------------------------------------------------------------------------
__global__ __launch_bounds__(256) void cvt_f32_f16(
    const float* __restrict__ src, _Float16* __restrict__ dst, int n4)
{
    int i = blockIdx.x * 256 + threadIdx.x;
    if (i < n4) {
        float4 a = ((const float4*)src)[i];
        v4h h = { (_Float16)a.x, (_Float16)a.y, (_Float16)a.z, (_Float16)a.w };
        ((v4h*)dst)[i] = h;
    }
}

// ---------------------------------------------------------------------------
// GEMM: out[M,N] = X[M,1024] * W[N,1024]^T + bias[N]   (M = 4096, f16 inputs)
// QKV mode scatters f16 q/k/v in [B,H,T,D]; else writes f32 row-major.
// Block tile 128x128, 8 waves (4x2), wave tile 32x64.
// K staged 64 at a time, double-buffered async DMA into LDS.
// ---------------------------------------------------------------------------
template <bool QKV, int N>
__global__ __launch_bounds__(256) void gemm_xwt(
    const _Float16* __restrict__ X, const _Float16* __restrict__ W,
    const float* __restrict__ bias, float* __restrict__ Cout,
    _Float16* __restrict__ qf, _Float16* __restrict__ kf, _Float16* __restrict__ vf)
{
    constexpr int Kd = CC;                // 1024
    __shared__ _Float16 As[2][128][72];   // 128 rows x 64 K, stride 72 (pad)
    __shared__ _Float16 Bs[2][128][72];

    const int tid  = threadIdx.x;
    const int wave = tid >> 5;
    const int lane = tid & 31;
    const int bm = blockIdx.y * 128;
    const int bn = blockIdx.x * 128;
    const int wm = (wave >> 1) * 32;   // 4 waves along M
    const int wn = (wave & 1) * 64;    // 2 waves along N

    auto stage = [&](int buf, int k0) {
#pragma unroll
        for (int i0 = 0; i0 < 4; ++i0) {
            int i = tid + i0 * 256;            // 1024 units of 16B
            int r = i >> 3, c8 = (i & 7) * 8;
            async_copy16(&X[(size_t)(bm + r) * Kd + k0 + c8], &As[buf][r][c8]);
            async_copy16(&W[(size_t)(bn + r) * Kd + k0 + c8], &Bs[buf][r][c8]);
        }
    };

    v8f acc[2][4] = {};

    stage(0, 0);
    async_wait0();
    __syncthreads();

    int buf = 0;
    for (int k0 = 0; k0 < Kd; k0 += 64, buf ^= 1) {
        if (k0 + 64 < Kd) stage(buf ^ 1, k0 + 64);   // prefetch next tile (DMA)

#pragma unroll
        for (int ks = 0; ks < 2; ++ks) {
            v16h af[2], bf[4];
#pragma unroll
            for (int i = 0; i < 2; ++i) af[i] = load_fragA(&As[buf][wm + 16 * i][ks * 32], 72, lane);
#pragma unroll
            for (int j = 0; j < 4; ++j) bf[j] = load_fragB(&Bs[buf][wn + 16 * j][ks * 32], 72, lane);
#pragma unroll
            for (int i = 0; i < 2; ++i)
#pragma unroll
                for (int j = 0; j < 4; ++j)
                    acc[i][j] = wmma16(af[i], bf[j], acc[i][j]);
        }
        async_wait0();
        __syncthreads();
    }

    const int roff = (lane & 16) ? 8 : 0;
    const int cl   = lane & 15;
#pragma unroll
    for (int i = 0; i < 2; ++i)
#pragma unroll
        for (int j = 0; j < 4; ++j) {
            int col = bn + wn + 16 * j + cl;
            float bv = bias[col];
#pragma unroll
            for (int r = 0; r < 8; ++r) {
                int row = bm + wm + 16 * i + r + roff;
                float val = acc[i][j][r] + bv;
                if (QKV) {
                    int b = row >> 11, t = row & (CT - 1);           // T = 2048
                    int mat = col >> 10, c = col & (CC - 1);         // C = 1024
                    int h = c >> 6, d = c & 63;                      // D = 64
                    size_t dst = ((size_t)((b << 4) + h) * CT + t) * CD + d;
                    _Float16 hv = (_Float16)val;
                    if (mat == 0) qf[dst] = hv;
                    else if (mat == 1) kf[dst] = hv;
                    else vf[dst] = hv;
                } else {
                    Cout[(size_t)row * N + col] = val;
                }
            }
        }
}

// ---------------------------------------------------------------------------
// Flash attention: Q,K,V f16 [B*H, T, 64] -> Y f16 [B, T, C]
// Block = 128 q rows of one (b,h); 8 waves, 16 q rows per wave.
// Keys streamed in chunks of 64 with online softmax; K/V double-buffered.
// ---------------------------------------------------------------------------
__global__ __launch_bounds__(256) void attn_kernel(
    const _Float16* __restrict__ Q, const _Float16* __restrict__ K,
    const _Float16* __restrict__ V, _Float16* __restrict__ Y)
{
    __shared__ _Float16 qs[128][72];      // 128 q rows x 64 d
    __shared__ _Float16 ks[2][64][72];    // 64 keys  x 64 d
    __shared__ _Float16 vts[2][64][72];   // V^T: 64 d x 64 keys
    __shared__ _Float16 ps[128][72];      // P: 128 q rows x 64 keys (wave-private rows)

    const int bh   = blockIdx.x;
    const int b    = bh >> 4;          // H = 16
    const int h    = bh & 15;
    const int q0   = blockIdx.y * 128;
    const int tid  = threadIdx.x;
    const int wave = tid >> 5;
    const int lane = tid & 31;

    const _Float16* Qb = Q + (size_t)bh * CT * CD;
    const _Float16* Kb = K + (size_t)bh * CT * CD;
    const _Float16* Vb = V + (size_t)bh * CT * CD;

    auto stage_kv = [&](int buf, int chunk) {
        const int kbase = chunk * 64;
#pragma unroll
        for (int i0 = 0; i0 < 2; ++i0) {
            int i = tid + i0 * 256;            // 512 units of 16B
            int r = i >> 3, c8 = (i & 7) * 8;
            async_copy16(&Kb[(size_t)(kbase + r) * CD + c8], &ks[buf][r][c8]);
            v8h vv = *(const v8h*)&Vb[(size_t)(kbase + r) * CD + c8];
#pragma unroll
            for (int e = 0; e < 8; ++e) vts[buf][c8 + e][r] = vv[e];
        }
    };

    // stage Q block (async DMA) + first K/V chunk
#pragma unroll
    for (int i0 = 0; i0 < 4; ++i0) {
        int i = tid + i0 * 256;
        int r = i >> 3, c8 = (i & 7) * 8;
        async_copy16(&Qb[(size_t)(q0 + r) * CD + c8], &qs[r][c8]);
    }
    stage_kv(0, 0);

    v8f o[4] = {};
    float mrow[8], lrow[8];
#pragma unroll
    for (int r = 0; r < 8; ++r) { mrow[r] = -1e30f; lrow[r] = 0.0f; }

    async_wait0();
    __syncthreads();

    const int myq   = q0 + wave * 16;
    const int roff  = (lane & 16) ? 8 : 0;
    const int mycol = lane & 15;
    const int nchunks = (q0 >> 6) + 2;   // keys needed up to q0+127

    const v16h aq0 = load_fragA(&qs[wave * 16][0], 72, lane);
    const v16h aq1 = load_fragA(&qs[wave * 16][32], 72, lane);

    for (int j = 0; j < nchunks; ++j) {
        const int buf   = j & 1;
        const int kbase = j * 64;
        if (j + 1 < nchunks) stage_kv(buf ^ 1, j + 1);   // prefetch next chunk

        // S = Q * K^T
        v8f s[4] = {};
#pragma unroll
        for (int kf2 = 0; kf2 < 4; ++kf2) {
            v16h bk0 = load_fragB(&ks[buf][kf2 * 16][0], 72, lane);
            v16h bk1 = load_fragB(&ks[buf][kf2 * 16][32], 72, lane);
            s[kf2] = wmma16(aq0, bk0, s[kf2]);
            s[kf2] = wmma16(aq1, bk1, s[kf2]);
        }

        // online softmax; rows r (lanes 0-15) / r+8 (lanes 16-31)
#pragma unroll
        for (int r = 0; r < 8; ++r) {
            int qrow = myq + r + roff;
            float sv[4];
            float cmax = -1e30f;
#pragma unroll
            for (int a = 0; a < 4; ++a) {
                int key = kbase + a * 16 + mycol;
                float val = s[a][r] * 0.125f;           // 1/sqrt(64)
                if (key > qrow) val = -1e30f;            // causal mask
                sv[a] = val;
                cmax = fmaxf(cmax, val);
            }
#pragma unroll
            for (int sft = 1; sft < 16; sft <<= 1)
                cmax = fmaxf(cmax, __shfl_xor(cmax, sft, 32));
            float newm  = fmaxf(mrow[r], cmax);
            float alpha = __expf(mrow[r] - newm);
            mrow[r] = newm;
            float rsum = 0.0f;
#pragma unroll
            for (int a = 0; a < 4; ++a) {
                float p = __expf(sv[a] - newm);
                rsum += p;
                ps[wave * 16 + r + roff][a * 16 + mycol] = (_Float16)p;
            }
#pragma unroll
            for (int sft = 1; sft < 16; sft <<= 1)
                rsum += __shfl_xor(rsum, sft, 32);
            lrow[r] = alpha * lrow[r] + rsum;
#pragma unroll
            for (int d = 0; d < 4; ++d) o[d][r] *= alpha;
        }

        // O += P * V   (P rows are wave-private in LDS; no barrier needed)
#pragma unroll
        for (int ks2 = 0; ks2 < 2; ++ks2) {
            v16h ap = load_fragA(&ps[wave * 16][ks2 * 32], 72, lane);
#pragma unroll
            for (int d = 0; d < 4; ++d) {
                v16h bv = load_fragB(&vts[buf][d * 16][ks2 * 32], 72, lane);
                o[d] = wmma16(ap, bv, o[d]);
            }
        }
        async_wait0();
        __syncthreads();   // next chunk's buffers ready; current buffers free
    }

    // write Y f16 [b, t, h*64 + d], normalized by row sum
#pragma unroll
    for (int r = 0; r < 8; ++r) {
        int t = myq + r + roff;
        float inv = 1.0f / lrow[r];
#pragma unroll
        for (int d = 0; d < 4; ++d) {
            int col = h * CD + d * 16 + mycol;
            Y[((size_t)b * CT + t) * CC + col] = (_Float16)(o[d][r] * inv);
        }
    }
}

// ---------------------------------------------------------------------------
extern "C" void kernel_launch(void* const* d_in, const int* in_sizes, int n_in,
                              void* d_out, int out_size, void* d_ws, size_t ws_size,
                              hipStream_t stream)
{
    const float* x      = (const float*)d_in[0];
    const float* w_attn = (const float*)d_in[1];
    const float* b_attn = (const float*)d_in[2];
    const float* w_proj = (const float*)d_in[3];
    const float* b_proj = (const float*)d_in[4];
    float* out = (float*)d_out;

    const int M = CB * CT;   // 4096

    // workspace: x16, wa16, wp16, q/k/v f16 [B,H,T,64], y16 [B,T,C]
    size_t xel  = (size_t)CB * CT * CC;       // 4M
    size_t wael = (size_t)3 * CC * CC;        // 3M
    size_t wpel = (size_t)CC * CC;            // 1M
    size_t qel  = (size_t)CB * CH * CT * CD;  // 4M

    _Float16* x16  = (_Float16*)d_ws;
    _Float16* wa16 = x16 + xel;
    _Float16* wp16 = wa16 + wael;
    _Float16* qf   = wp16 + wpel;
    _Float16* kf   = qf + qel;
    _Float16* vf   = kf + qel;
    _Float16* y16  = vf + qel;

    dim3 blk(256);

    // f32 -> f16 conversions (bandwidth-trivial one-shot)
    cvt_f32_f16<<<dim3((int)(xel / 4 / 256)), blk, 0, stream>>>(x, x16, (int)(xel / 4));
    cvt_f32_f16<<<dim3((int)(wael / 4 / 256)), blk, 0, stream>>>(w_attn, wa16, (int)(wael / 4));
    cvt_f32_f16<<<dim3((int)(wpel / 4 / 256)), blk, 0, stream>>>(w_proj, wp16, (int)(wpel / 4));

    // QKV projection: [M, 3C] = x @ w_attn^T + b_attn, scattered to q/k/v f16
    gemm_xwt<true, 3 * CC><<<dim3(3 * CC / 128, M / 128), blk, 0, stream>>>(
        x16, wa16, b_attn, nullptr, qf, kf, vf);

    // causal flash attention -> y16
    attn_kernel<<<dim3(CB * CH, CT / 128), blk, 0, stream>>>(qf, kf, vf, y16);

    // output projection: out = y @ w_proj^T + b_proj
    gemm_xwt<false, CC><<<dim3(CC / 128, M / 128), blk, 0, stream>>>(
        y16, wp16, b_proj, out, nullptr, nullptr, nullptr);
}
